// m2oAttentionA3_32177894982193
// MI455X (gfx1250) — matrix operand order repaired and checked by reading, not compiled, not run on hardware
//
#include <hip/hip_runtime.h>

// ---------- types ----------
typedef __attribute__((ext_vector_type(16))) __bf16 bf16x16;
typedef __attribute__((ext_vector_type(8)))  float  floatx8;
typedef int v4i32 __attribute__((vector_size(16)));

__device__ __forceinline__ float fast_tanh(float x) {
#if __has_builtin(__builtin_amdgcn_tanhf)
  return __builtin_amdgcn_tanhf(x);
#elif __has_builtin(__builtin_amdgcn_tanh_f32)
  return __builtin_amdgcn_tanh_f32(x);
#else
  return tanhf(x);
#endif
}

// ---------- async Global->LDS 16B copy (CDNA5 GLOBAL_LOAD_ASYNC_TO_LDS_B128) ----------
#if __has_builtin(__builtin_amdgcn_global_load_async_to_lds_b128) && \
    __has_builtin(__builtin_amdgcn_s_wait_asynccnt)
#define HAVE_ASYNC_LDS 1
#endif

__device__ __forceinline__ void copy16_to_lds(void* lds, const void* gmem) {
#ifdef HAVE_ASYNC_LDS
  __builtin_amdgcn_global_load_async_to_lds_b128(
      (v4i32*)const_cast<void*>(gmem), (v4i32*)lds, 0, 0);
#else
  *(float4*)lds = *(const float4*)gmem;
#endif
}
__device__ __forceinline__ void copy_join() {
#ifdef HAVE_ASYNC_LDS
  __builtin_amdgcn_s_wait_asynccnt(0);
#endif
}

// =====================================================================
// Generic GEMM: C[M,N] = A[M,K] @ B[K,N] (+ bias[N]) (+ addmat[M,N]) (tanh)
// f32 in/out, bf16 WMMA compute, f32 LDS tiles staged with async b128 copies.
// Block = 128 threads = 4 waves. Block tile 64(M) x 64(N); wave w owns rows
// [wm, wm+16) x all 64 cols. Requires M%64==0, K%32==0, N%64==0.
// =====================================================================
__global__ __launch_bounds__(128)
void gemm_bf16_wmma(const float* __restrict__ A, const float* __restrict__ Bm,
                    const float* __restrict__ bias, const float* __restrict__ addmat,
                    float* __restrict__ C, int M, int K, int N, int do_tanh)
{
  __shared__ float As[64][32];     // [m][k]
  __shared__ float Bsf[32][64];    // [k][n]
  const int tid  = threadIdx.x;
  const int wave = tid >> 5;
  const int lane = tid & 31;
  const int m0 = blockIdx.x * 64;
  const int n0 = blockIdx.y * 64;
  const int wm = m0 + wave * 16;

  floatx8 acc[4] = {};

  for (int k0 = 0; k0 < K; k0 += 32) {
    // stage A tile: 64x32 f32 = 512 x 16B chunks, 4 per thread
#pragma unroll
    for (int i = 0; i < 4; ++i) {
      int c = tid + i * 128;
      int r = c >> 3, g = (c & 7) * 4;
      copy16_to_lds(&As[r][g], &A[(size_t)(m0 + r) * K + (k0 + g)]);
    }
    // stage B tile: 32x64 f32 = 512 x 16B chunks (n-contiguous, coalesced)
#pragma unroll
    for (int i = 0; i < 4; ++i) {
      int c = tid + i * 128;
      int kk = c >> 4, g = (c & 15) * 4;
      copy16_to_lds(&Bsf[kk][g], &Bm[(size_t)(k0 + kk) * N + (n0 + g)]);
    }
    copy_join();
    __syncthreads();

    // A fragment (16x32 bf16): lane<16 -> K 0..15 ; lane>=16 -> K 16..31
    bf16x16 afrag;
    {
      const int mr = wave * 16 + (lane & 15);
      const int kadd = (lane < 16) ? 0 : 16;
#pragma unroll
      for (int v = 0; v < 8; ++v) {
        int kb = 2 * v + kadd;
        afrag[2 * v]     = (__bf16)As[mr][kb];
        afrag[2 * v + 1] = (__bf16)As[mr][kb + 1];
      }
    }
#pragma unroll
    for (int ns = 0; ns < 4; ++ns) {
      // B fragment (32x16): lane<16 -> K 0..15 ; lane>=16 -> K 16..31
      bf16x16 bfrag;
      const int nn = ns * 16 + (lane & 15);
      const int kadd = (lane < 16) ? 0 : 16;
#pragma unroll
      for (int v = 0; v < 8; ++v) {
        bfrag[2 * v]     = (__bf16)Bsf[2 * v + kadd][nn];
        bfrag[2 * v + 1] = (__bf16)Bsf[2 * v + kadd + 1][nn];
      }
      acc[ns] = __builtin_amdgcn_wmma_f32_16x16x32_bf16(
          false, afrag, false, bfrag, (short)0, acc[ns], false, false);
    }
    __syncthreads();
  }

  // epilogue: C layout VGPR r -> M = r (+8 for lanes 16..31), N = lane&15
#pragma unroll
  for (int ns = 0; ns < 4; ++ns) {
    const int nn = n0 + ns * 16 + (lane & 15);
    const float bv = bias ? bias[nn] : 0.0f;
#pragma unroll
    for (int r = 0; r < 8; ++r) {
      const int mm = wm + r + ((lane < 16) ? 0 : 8);
      if (mm < M) {
        float v = acc[ns][r] + bv;
        if (addmat) v += addmat[(size_t)mm * N + nn];
        if (do_tanh) v = fast_tanh(v);
        C[(size_t)mm * N + nn] = v;
      }
    }
  }
}

// =====================================================================
// Scores + softmax. One wave per (b,s). Lane l covers H channels l+32j.
// sc[k] = V . tanh(X1h[(25b+k)%B, s, :] + X2h[b,k,:]) + bV ; softmax over k.
// mask==0 -> all scores -1e10 -> uniform 1/25.
// =====================================================================
__global__ __launch_bounds__(128)
void attn_scores_softmax(const float* __restrict__ X1h, const float* __restrict__ X2h,
                         const int* __restrict__ mask, const float* __restrict__ V,
                         const float* __restrict__ bV, float* __restrict__ att,
                         int Bsz, int S, int H)
{
  const int wave = threadIdx.x >> 5;
  const int lane = threadIdx.x & 31;
  const int idx = blockIdx.x * 4 + wave;
  if (idx >= Bsz * S) return;
  const int b = idx / S;
  const int s = idx - b * S;

  float vv[8];
#pragma unroll
  for (int j = 0; j < 8; ++j) vv[j] = V[lane + 32 * j];
  const float bVv = bV[0];

  float sc[25];
#pragma unroll 1
  for (int k = 0; k < 25; ++k) {
    const int bp = (b * 25 + k) % Bsz;              // torch repeat/view quirk
    const float* x1 = X1h + ((size_t)bp * S + s) * H;
    const float* x2 = X2h + ((size_t)(b * 25 + k)) * H;
    if (k < 24) {                                    // prefetch next candidate's row
      const int bpn = (b * 25 + k + 1) % Bsz;
      __builtin_prefetch(X1h + ((size_t)bpn * S + s) * H + lane, 0, 0);
    }
    float acc = 0.f;
#pragma unroll
    for (int j = 0; j < 8; ++j) {
      const int h = lane + 32 * j;
      acc += fast_tanh(x1[h] + x2[h]) * vv[j];
    }
#pragma unroll
    for (int off = 16; off > 0; off >>= 1)
      acc += __shfl_xor(acc, off, 32);
    sc[k] = acc + bVv;
  }

  float w[25];
  if (mask[b * S + s] == 0) {
#pragma unroll
    for (int k = 0; k < 25; ++k) w[k] = 1.0f / 25.0f;
  } else {
    float mx = sc[0];
#pragma unroll
    for (int k = 1; k < 25; ++k) mx = fmaxf(mx, sc[k]);
    float sum = 0.f;
#pragma unroll
    for (int k = 0; k < 25; ++k) { w[k] = __expf(sc[k] - mx); sum += w[k]; }
    const float inv = 1.0f / sum;
#pragma unroll
    for (int k = 0; k < 25; ++k) w[k] *= inv;
  }
  if (lane == 0) {
#pragma unroll
    for (int k = 0; k < 25; ++k)
      att[((size_t)b * 25 + k) * S + s] = w[k];
  }
}

// =====================================================================
// Per-batch weighted sum via WMMA: out[b] = att[b](25x100) @ inp1[b](100x256)
// M padded 25->32, K padded 100->128 with zeros. Block = 256 thr = 8 waves,
// wave w owns 32 output cols; acc[2][2] 16x16 tiles. f32 LDS tiles, async
// staging (zero-pad chunks stay aligned since S=100 is a multiple of 4).
// =====================================================================
__global__ __launch_bounds__(256)
void weighted_sum_wmma(const float* __restrict__ att, const float* __restrict__ inp1,
                       float* __restrict__ out, int S, int D)
{
  __shared__ float Asf[32][32];      // [m][k] step tile
  __shared__ float Bsf[32][256];     // [k][n]
  const int b = blockIdx.x;
  const int tid = threadIdx.x, wave = tid >> 5, lane = tid & 31;
  const float* attb = att + (size_t)b * 25 * S;
  const float* x1b  = inp1 + (size_t)b * S * D;

  floatx8 acc[2][2] = {};

  for (int k0 = 0; k0 < 128; k0 += 32) {
    // stage A (att rows): 32x32 f32 = 256 x 16B chunks, 1 per thread
    {
      int r = tid >> 3, g = (tid & 7) * 4;
      if (r < 25 && (k0 + g) < S) {
        copy16_to_lds(&Asf[r][g], &attb[r * S + k0 + g]);
      } else {
        Asf[r][g] = 0.f; Asf[r][g + 1] = 0.f; Asf[r][g + 2] = 0.f; Asf[r][g + 3] = 0.f;
      }
    }
    // stage B (inp1 rows k0..k0+31 x 256 cols): 2048 x 16B chunks, 8 per thread
#pragma unroll
    for (int i = 0; i < 8; ++i) {
      int c = tid + i * 256;
      int kk = c >> 6, g = (c & 63) * 4;
      if ((k0 + kk) < S) {
        copy16_to_lds(&Bsf[kk][g], &x1b[(size_t)(k0 + kk) * D + g]);
      } else {
        Bsf[kk][g] = 0.f; Bsf[kk][g + 1] = 0.f; Bsf[kk][g + 2] = 0.f; Bsf[kk][g + 3] = 0.f;
      }
    }
    copy_join();
    __syncthreads();

#pragma unroll
    for (int mt = 0; mt < 2; ++mt) {
      bf16x16 afrag;
      const int mr = mt * 16 + (lane & 15);
      const int kadd = (lane < 16) ? 0 : 16;
#pragma unroll
      for (int v = 0; v < 8; ++v) {
        int kb = 2 * v + kadd;
        afrag[2 * v]     = (__bf16)Asf[mr][kb];
        afrag[2 * v + 1] = (__bf16)Asf[mr][kb + 1];
      }
#pragma unroll
      for (int nt = 0; nt < 2; ++nt) {
        bf16x16 bfrag;
        const int nn = wave * 32 + nt * 16 + (lane & 15);
        const int kadd2 = (lane < 16) ? 0 : 16;
#pragma unroll
        for (int v = 0; v < 8; ++v) {
          bfrag[2 * v]     = (__bf16)Bsf[2 * v + kadd2][nn];
          bfrag[2 * v + 1] = (__bf16)Bsf[2 * v + kadd2 + 1][nn];
        }
        acc[mt][nt] = __builtin_amdgcn_wmma_f32_16x16x32_bf16(
            false, afrag, false, bfrag, (short)0, acc[mt][nt], false, false);
      }
    }
    __syncthreads();
  }

#pragma unroll
  for (int mt = 0; mt < 2; ++mt)
#pragma unroll
    for (int nt = 0; nt < 2; ++nt) {
      const int nn = wave * 32 + nt * 16 + (lane & 15);
#pragma unroll
      for (int r = 0; r < 8; ++r) {
        const int mm = mt * 16 + r + ((lane < 16) ? 0 : 8);
        if (mm < 25) out[((size_t)b * 25 + mm) * D + nn] = acc[mt][nt][r];
      }
    }
}

// =====================================================================
// Host-side launch
// =====================================================================
extern "C" void kernel_launch(void* const* d_in, const int* in_sizes, int n_in,
                              void* d_out, int out_size, void* d_ws, size_t ws_size,
                              hipStream_t stream) {
  (void)in_sizes; (void)n_in; (void)out_size; (void)ws_size;
  const int B = 128, S = 100, D1 = 256, H = 256, NC = 25;

  const float* inp1 = (const float*)d_in[0];   // [B,S,D1]
  const float* inp2 = (const float*)d_in[1];   // [B,25,D2]
  const int*   mask = (const int*)  d_in[2];   // [B,S]
  const float* W1   = (const float*)d_in[3];
  const float* W2   = (const float*)d_in[4];
  const float* b2   = (const float*)d_in[5];
  const float* W3   = (const float*)d_in[6];
  const float* W4   = (const float*)d_in[7];
  const float* b4   = (const float*)d_in[8];
  const float* V    = (const float*)d_in[9];   // [H,1]
  const float* bV   = (const float*)d_in[10];  // [1]
  float* out = (float*)d_out;                  // [B,25,H]

  // workspace partition (floats)
  float* ws   = (float*)d_ws;
  float* X1h  = ws;                                  // B*S*H     = 3,276,800
  float* X2h  = X1h + (size_t)B * S * H;             // B*25*H    =   819,200
  float* X4   = X2h + (size_t)B * NC * H;            // B*25*H    =   819,200
  float* attw = X4  + (size_t)B * NC * H;            // B*25*S    =   320,000
  float* iav  = attw + (size_t)B * NC * S;           // B*25*D1   =   819,200

  const int M1 = B * S;    // 12800
  const int M2 = B * NC;   // 3200

  // 1) X1h = inp1 @ W1
  gemm_bf16_wmma<<<dim3(M1 / 64, H / 64), 128, 0, stream>>>(
      inp1, W1, nullptr, nullptr, X1h, M1, D1, H, 0);
  // 2) X2h = inp2 @ W2 + b2
  gemm_bf16_wmma<<<dim3(M2 / 64, H / 64), 128, 0, stream>>>(
      inp2, W2, b2, nullptr, X2h, M2, D1, H, 0);
  // 3) X4 = inp2 @ W4 + b4
  gemm_bf16_wmma<<<dim3(M2 / 64, H / 64), 128, 0, stream>>>(
      inp2, W4, b4, nullptr, X4, M2, D1, H, 0);
  // 4) scores + softmax -> attw
  attn_scores_softmax<<<(B * S) / 4, 128, 0, stream>>>(
      X1h, X2h, mask, V, bV, attw, B, S, H);
  // 5) iav[b] = attw[b] @ inp1[b]
  weighted_sum_wmma<<<B, 256, 0, stream>>>(attw, inp1, iav, S, D1);
  // 6) out = tanh(iav @ W3 + X4)
  gemm_bf16_wmma<<<dim3(M2 / 64, H / 64), 128, 0, stream>>>(
      iav, W3, nullptr, X4, out, M2, D1, H, 1);
}
